// QWenAttention_6682969113256
// MI455X (gfx1250) — matrix operand order repaired and checked by reading, not compiled, not run on hardware
//
#include <hip/hip_runtime.h>
#include <cstddef>
#include <cstdint>

#define Bx 2
#define Sx 2048
#define Hx 4096
#define NHx 32
#define HDx 128
#define HALF 64

typedef __attribute__((ext_vector_type(16))) _Float16 v16h;
typedef __attribute__((ext_vector_type(8)))  _Float16 v8h;
typedef __attribute__((ext_vector_type(4)))  _Float16 v4h;
typedef __attribute__((ext_vector_type(8)))  float    v8f;

// Parameter type expected by __builtin_amdgcn_global_load_async_to_lds_b128
typedef int v4i_vs __attribute__((vector_size(16)));

#if __has_builtin(__builtin_amdgcn_global_load_async_to_lds_b128)
#define HAVE_ASYNC_LDS 1
#else
#define HAVE_ASYNC_LDS 0
#endif

__device__ __forceinline__ void async_copy16(const _Float16* g, _Float16* l) {
#if HAVE_ASYNC_LDS
    __builtin_amdgcn_global_load_async_to_lds_b128(
        (v4i_vs*)g, (v4i_vs*)l, 0, 0);
#else
    *(v8h*)l = *(const v8h*)g;
#endif
}

__device__ __forceinline__ void wait_async() {
#if HAVE_ASYNC_LDS
#if __has_builtin(__builtin_amdgcn_s_wait_asynccnt)
    __builtin_amdgcn_s_wait_asynccnt(0);
#else
    asm volatile("s_wait_asynccnt 0x0" ::: "memory");
#endif
#endif
}

__device__ __forceinline__ v8f wmma_f16(v16h a, v16h b, v8f c) {
    // D = A(16x32 f16) * B(32x16 f16) + C(16x16 f32)
    return __builtin_amdgcn_wmma_f32_16x16x32_f16(
        false, a, false, b, (short)0, c, false, false);
}

__device__ __forceinline__ v16h pack16(const _Float16* lo, const _Float16* hi) {
    v8h a = *(const v8h*)lo;
    v8h b = *(const v8h*)hi;
    v16h r;
#pragma unroll
    for (int i = 0; i < 8; ++i) { r[i] = a[i]; r[i + 8] = b[i]; }
    return r;
}

// ---------------------------------------------------------------------------
// Elementwise f32 -> f16 (hidden states). 4 elements/thread.
// ---------------------------------------------------------------------------
__global__ void conv_f32_f16(const float* __restrict__ src, _Float16* __restrict__ dst)
{
    size_t i = ((size_t)blockIdx.x * blockDim.x + threadIdx.x) * 4;
    float4 f = *(const float4*)(src + i);
    v4h d;
    d[0] = (_Float16)f.x; d[1] = (_Float16)f.y;
    d[2] = (_Float16)f.z; d[3] = (_Float16)f.w;
    *(v4h*)(dst + i) = d;
}

// ---------------------------------------------------------------------------
// W[K][N] f32 -> Wt[N][K] f16, one 64x64 tile per 256-thread block.
// ---------------------------------------------------------------------------
__global__ __launch_bounds__(256)
void transpose_f32_f16(const float* __restrict__ W, _Float16* __restrict__ Wt,
                       int K, int N)
{
    __shared__ __align__(16) _Float16 t[64][80];    // stride 160B (16B multiple)
    const int tid = threadIdx.x;
    const int k0 = blockIdx.x * 64;
    const int n0 = blockIdx.y * 64;
#pragma unroll
    for (int p = 0; p < 4; ++p) {
        int linear = p * 1024 + tid * 4;
        int r = linear >> 6, c = linear & 63;       // r: k offset, c: n offset
        float4 f = *(const float4*)(W + (size_t)(k0 + r) * N + n0 + c);
        t[c + 0][r] = (_Float16)f.x; t[c + 1][r] = (_Float16)f.y;
        t[c + 2][r] = (_Float16)f.z; t[c + 3][r] = (_Float16)f.w;
    }
    __syncthreads();
#pragma unroll
    for (int p = 0; p < 2; ++p) {
        int linear = p * 2048 + tid * 8;
        int r = linear >> 6, c = linear & 63;       // r: n offset, c: k offset
        v8h d = *(const v8h*)&t[r][c];
        *(v8h*)(Wt + (size_t)(n0 + r) * K + k0 + c) = d;
    }
}

// ---------------------------------------------------------------------------
// GEMM: C[M,N] = A16[M,K] @ Wt16[N,K]^T (+ bias). f16 WMMA, f32 accum/out.
// 128x128 tile per 256-thread block (8 waves); wave = 32x64 subtile.
// K staged 64 at a time, double-buffered pure-async LDS staging.
// ---------------------------------------------------------------------------
__global__ __launch_bounds__(256)
void gemm16_bias_wmma(const _Float16* __restrict__ A, const _Float16* __restrict__ Wt,
                      const float* __restrict__ bias, float* __restrict__ C,
                      int M, int N, int K)
{
    __shared__ __align__(16) _Float16 As[2][128][64];  // [buf][m][k]
    __shared__ __align__(16) _Float16 Ws[2][128][64];  // [buf][n][k]

    const int tid   = threadIdx.x;
    const int lane  = tid & 31;
    const int wave  = tid >> 5;
    const int wm    = wave & 3;     // M offset wm*32
    const int wn    = wave >> 2;    // N offset wn*64
    const int lhalf = lane & 15;
    const int hsel  = lane >> 4;    // 0 or 1
    const int bm = blockIdx.x * 128;
    const int bn = blockIdx.y * 128;

    v8f acc[2][4];
#pragma unroll
    for (int i = 0; i < 2; ++i)
#pragma unroll
        for (int j = 0; j < 4; ++j)
#pragma unroll
            for (int r = 0; r < 8; ++r) acc[i][j][r] = 0.0f;

    auto stage = [&](int s, int buf) {
        const int k0 = s * 64;
        // A tile: 128 rows x 64 halves = 1024 x 16B chunks
#pragma unroll
        for (int p = 0; p < 4; ++p) {
            int cz = p * 256 + tid;
            int r = cz >> 3, c8 = (cz & 7) * 8;
            async_copy16(A + (size_t)(bm + r) * K + k0 + c8, &As[buf][r][c8]);
        }
        // W tile (already [N][K]): 128 rows x 64 halves
#pragma unroll
        for (int p = 0; p < 4; ++p) {
            int cz = p * 256 + tid;
            int r = cz >> 3, c8 = (cz & 7) * 8;
            async_copy16(Wt + (size_t)(bn + r) * K + k0 + c8, &Ws[buf][r][c8]);
        }
    };

    const int nst = K / 64;
    stage(0, 0);
    for (int s = 0; s < nst; ++s) {
        const int buf = s & 1;
        wait_async();
        __syncthreads();
        if (s + 1 < nst) stage(s + 1, (s + 1) & 1);

#pragma unroll
        for (int kk = 0; kk < 2; ++kk) {
            v16h af[2];
#pragma unroll
            for (int fm = 0; fm < 2; ++fm) {
                const _Float16* row = &As[buf][wm * 32 + fm * 16 + lhalf][kk * 32];
                int kA = hsel * 8;      // lanes0-15: K{0..7,16..23}; 16-31: {8..15,24..31}
                af[fm] = pack16(row + kA, row + kA + 16);
            }
#pragma unroll
            for (int fn = 0; fn < 4; ++fn) {
                const _Float16* row = &Ws[buf][wn * 64 + fn * 16 + lhalf][kk * 32];
                int kB = hsel * 8;      // lanes0-15: K{0..7,16..23}; 16-31: {8..15,24..31}
                v16h bf = pack16(row + kB, row + kB + 16);
#pragma unroll
                for (int fm = 0; fm < 2; ++fm)
                    acc[fm][fn] = wmma_f16(af[fm], bf, acc[fm][fn]);
            }
        }
    }

#pragma unroll
    for (int fm = 0; fm < 2; ++fm)
#pragma unroll
        for (int fn = 0; fn < 4; ++fn) {
            int gn = bn + wn * 64 + fn * 16 + lhalf;
            float bv = bias ? bias[gn] : 0.0f;
#pragma unroll
            for (int r = 0; r < 8; ++r) {
                int gm = bm + wm * 32 + fm * 16 + r + hsel * 8;
                C[(size_t)gm * N + gn] = acc[fm][fn][r] + bv;
            }
        }
}

// ---------------------------------------------------------------------------
// RoPE: reads raw fp32 q,k and fp32 v (already in d_out). Emits:
//   outK (fp32 roped k, d_out region), and f16 shadows q16/k16/v16 for attn.
// One thread per (b,s,head,i<64) rotation pair.
// ---------------------------------------------------------------------------
__global__ void rope_kernel(const int* __restrict__ pos,
                            const float* __restrict__ qraw,
                            const float* __restrict__ kraw,
                            const float* __restrict__ vout,
                            float* __restrict__ outK,
                            _Float16* __restrict__ q16,
                            _Float16* __restrict__ k16,
                            _Float16* __restrict__ v16)
{
    size_t idx = (size_t)blockIdx.x * blockDim.x + threadIdx.x;
    const size_t total = (size_t)Bx * Sx * NHx * HALF;
    if (idx >= total) return;
    int i = (int)(idx % HALF);
    size_t t = idx / HALF;
    int h = (int)(t % NHx);
    size_t bs = t / NHx;

    float p = (float)pos[bs];
    float inv = __expf(-(2.0f * (float)i / (float)HDx) * 9.210340371976184f);
    float f = p * inv;
    float s, c;
    __sincosf(f, &s, &c);

    size_t base = bs * Hx + (size_t)h * HDx + i;
    float q1 = qraw[base], q2 = qraw[base + HALF];
    q16[base]        = (_Float16)(q1 * c - q2 * s);
    q16[base + HALF] = (_Float16)(q2 * c + q1 * s);

    float k1 = kraw[base], k2 = kraw[base + HALF];
    float kr1 = k1 * c - k2 * s, kr2 = k2 * c + k1 * s;
    outK[base] = kr1;             outK[base + HALF] = kr2;
    k16[base]  = (_Float16)kr1;   k16[base + HALF]  = (_Float16)kr2;

    v16[base]        = (_Float16)vout[base];
    v16[base + HALF] = (_Float16)vout[base + HALF];
}

// ---------------------------------------------------------------------------
// Causal flash attention over f16 q/k/v. Block = (b, h, 128 q-rows),
// 8 waves x 16 rows each. Q/K tiles via async LDS copies; V transposed via
// VGPR path. Scores + PV in f16 WMMA, fp32 online softmax. Output f16.
// ---------------------------------------------------------------------------
__global__ __launch_bounds__(256)
void attn_kernel(const _Float16* __restrict__ Q, const _Float16* __restrict__ Kc,
                 const _Float16* __restrict__ V, _Float16* __restrict__ Oa16)
{
    __shared__ __align__(16) _Float16 Qs[128][HDx]; // [qrow][hd]
    __shared__ __align__(16) _Float16 Ks[32][HDx];  // [kidx][hd]
    __shared__ __align__(16) _Float16 Vs[HDx][32];  // [hd][kidx] (transposed)
    __shared__ __align__(16) _Float16 Ps[128][32];  // per-wave 16x32 prob tiles

    const int tid   = threadIdx.x;
    const int lane  = tid & 31;
    const int wave  = tid >> 5;
    const int lhalf = lane & 15;
    const int hsel  = lane >> 4;

    const int nq = Sx / 128;
    int qb = blockIdx.x % nq;
    int h  = (blockIdx.x / nq) % NHx;
    int b  = blockIdx.x / (nq * NHx);
    const int q0 = qb * 128;

    const size_t qgbase = ((size_t)b * Sx + q0) * Hx + (size_t)h * HDx;
#pragma unroll
    for (int p = 0; p < 8; ++p) {
        int cz = p * 256 + tid;
        int r = cz >> 4, c8 = (cz & 15) * 8;
        async_copy16(Q + qgbase + (size_t)r * Hx + c8, &Qs[r][c8]);
    }
    wait_async();
    __syncthreads();

    v16h qf[4];
#pragma unroll
    for (int kk = 0; kk < 4; ++kk) {
        const _Float16* row = &Qs[wave * 16 + lhalf][kk * 32];
        int kA = hsel * 8;
        qf[kk] = pack16(row + kA, row + kA + 16);
    }

    v8f o[8];
    float mrow[8], lrow[8];
#pragma unroll
    for (int d = 0; d < 8; ++d)
#pragma unroll
        for (int r = 0; r < 8; ++r) o[d][r] = 0.0f;
#pragma unroll
    for (int r = 0; r < 8; ++r) { mrow[r] = -3.0e38f; lrow[r] = 0.0f; }

    const float scale = 0.08838834764831845f; // 1/sqrt(128)
    const int nkv = (q0 + 127) / 32 + 1;      // causal tile count

    for (int j = 0; j < nkv; ++j) {
        const int kbase = j * 32;
        const size_t kgbase = ((size_t)b * Sx + kbase) * Hx + (size_t)h * HDx;
#pragma unroll
        for (int p = 0; p < 2; ++p) {
            int cz = p * 256 + tid;
            int r = cz >> 4, c8 = (cz & 15) * 8;
            async_copy16(Kc + kgbase + (size_t)r * Hx + c8, &Ks[r][c8]);
        }
#pragma unroll
        for (int p = 0; p < 2; ++p) {
            int cz = p * 256 + tid;
            int r = cz >> 4, c8 = (cz & 15) * 8;
            v8h d = *(const v8h*)(V + kgbase + (size_t)r * Hx + c8);
#pragma unroll
            for (int i = 0; i < 8; ++i) Vs[c8 + i][r] = d[i];
        }
        wait_async();
        __syncthreads();

        v8f sf[2];
#pragma unroll
        for (int n = 0; n < 2; ++n) {
#pragma unroll
            for (int r = 0; r < 8; ++r) sf[n][r] = 0.0f;
#pragma unroll
            for (int kk = 0; kk < 4; ++kk) {
                const _Float16* row = &Ks[n * 16 + lhalf][kk * 32 + hsel * 8];
                v16h bf = pack16(row, row + 16);
                sf[n] = wmma_f16(qf[kk], bf, sf[n]);
            }
        }

#pragma unroll
        for (int r = 0; r < 8; ++r) {
            int qrow = q0 + wave * 16 + r + hsel * 8;
#pragma unroll
            for (int n = 0; n < 2; ++n) {
                int kpos = kbase + n * 16 + lhalf;
                float sv = sf[n][r] * scale;
                sf[n][r] = (kpos <= qrow) ? sv : -3.0e38f;
            }
            float t = fmaxf(sf[0][r], sf[1][r]);
#pragma unroll
            for (int off = 1; off < 16; off <<= 1)
                t = fmaxf(t, __shfl_xor(t, off, 16));
            float mnew = fmaxf(mrow[r], t);
            float alpha = __expf(mrow[r] - mnew);
            mrow[r] = mnew;
            float psum = 0.0f;
#pragma unroll
            for (int n = 0; n < 2; ++n) {
                float pv = __expf(sf[n][r] - mnew);
                sf[n][r] = pv;
                psum += pv;
            }
#pragma unroll
            for (int off = 1; off < 16; off <<= 1)
                psum += __shfl_xor(psum, off, 16);
            lrow[r] = lrow[r] * alpha + psum;
#pragma unroll
            for (int d = 0; d < 8; ++d) o[d][r] *= alpha;
            int prow = wave * 16 + r + hsel * 8;
            Ps[prow][lhalf]      = (_Float16)sf[0][r];
            Ps[prow][16 + lhalf] = (_Float16)sf[1][r];
        }

        {
            const _Float16* prow = &Ps[wave * 16 + lhalf][0];
            int kA = hsel * 8;
            v16h pf = pack16(prow + kA, prow + kA + 16);
#pragma unroll
            for (int d = 0; d < 8; ++d) {
                const _Float16* vrow = &Vs[d * 16 + lhalf][hsel * 8];
                v16h vf = pack16(vrow, vrow + 16);
                o[d] = wmma_f16(pf, vf, o[d]);
            }
        }
        __syncthreads();
    }

#pragma unroll
    for (int d = 0; d < 8; ++d)
#pragma unroll
        for (int r = 0; r < 8; ++r) {
            int qrow = q0 + wave * 16 + r + hsel * 8;
            float val = o[d][r] / lrow[r];
            Oa16[((size_t)b * Sx + qrow) * Hx + (size_t)h * HDx + d * 16 + lhalf] =
                (_Float16)val;
        }
}

// ---------------------------------------------------------------------------
extern "C" void kernel_launch(void* const* d_in, const int* in_sizes, int n_in,
                              void* d_out, int out_size, void* d_ws, size_t ws_size,
                              hipStream_t stream) {
    const int*   positions = (const int*)d_in[0];
    const float* hidden    = (const float*)d_in[1];
    const float* Wq        = (const float*)d_in[2];
    const float* bq        = (const float*)d_in[3];
    const float* Wk        = (const float*)d_in[4];
    const float* bk        = (const float*)d_in[5];
    const float* Wv        = (const float*)d_in[6];
    const float* bv        = (const float*)d_in[7];
    const float* Wo        = (const float*)d_in[8];

    const size_t Nf = (size_t)Bx * Sx * Hx;   // 16,777,216 elements per tensor

    float*    ws_qraw = (float*)d_ws;              // raw q (fp32)
    float*    ws_kraw = ws_qraw + Nf;              // raw k (fp32)
    _Float16* q16     = (_Float16*)(ws_kraw + Nf);
    _Float16* k16     = q16 + Nf;
    _Float16* v16     = k16 + Nf;
    _Float16* a16     = v16 + Nf;                  // hidden16, later attn16
    _Float16* w16t    = a16 + Nf;                  // current weight, [N][K] f16

    float* out  = (float*)d_out;        // [B,S,H] final
    float* outK = out + Nf;             // roped k (fp32)
    float* outV = outK + Nf;            // v (fp32) -- written directly by GEMM

    const int M = Bx * Sx;              // 4096
    dim3 gGrid(M / 128, Hx / 128);      // 32 x 32
    dim3 tGrid(Hx / 64, Hx / 64);       // 64 x 64 transpose tiles
    dim3 blk(256);
    const unsigned convBlocks = (unsigned)(Nf / 4 / 256); // 16384

    conv_f32_f16<<<convBlocks, blk, 0, stream>>>(hidden, a16);

    transpose_f32_f16<<<tGrid, blk, 0, stream>>>(Wq, w16t, Hx, Hx);
    gemm16_bias_wmma<<<gGrid, blk, 0, stream>>>(a16, w16t, bq, ws_qraw, M, Hx, Hx);
    transpose_f32_f16<<<tGrid, blk, 0, stream>>>(Wk, w16t, Hx, Hx);
    gemm16_bias_wmma<<<gGrid, blk, 0, stream>>>(a16, w16t, bk, ws_kraw, M, Hx, Hx);
    transpose_f32_f16<<<tGrid, blk, 0, stream>>>(Wv, w16t, Hx, Hx);
    gemm16_bias_wmma<<<gGrid, blk, 0, stream>>>(a16, w16t, bv, outV, M, Hx, Hx);

    const size_t rope_total = (size_t)Bx * Sx * NHx * HALF; // 8,388,608
    rope_kernel<<<(unsigned)((rope_total + 255) / 256), blk, 0, stream>>>(
        positions, ws_qraw, ws_kraw, outV, outK, q16, k16, v16);

    dim3 aGrid((unsigned)(Bx * NHx * (Sx / 128)));          // 1024
    attn_kernel<<<aGrid, blk, 0, stream>>>(q16, k16, v16, a16);  // a16 reused as attn16

    transpose_f32_f16<<<tGrid, blk, 0, stream>>>(Wo, w16t, Hx, Hx);
    gemm16_bias_wmma<<<gGrid, blk, 0, stream>>>(a16, w16t, nullptr, out, M, Hx, Hx);
}
